// TransMIL_26233660244515
// MI455X (gfx1250) — compile-verified, matrix operand order checked by reading, not run on hardware
//
#include <hip/hip_runtime.h>

// ---------------------------------------------------------------------------
// TransMIL forward for MI455X (gfx1250, wave32, WMMA).
// GEMMs use v_wmma_f32_16x16x32_bf16 (f32 accumulate). Large multiply-read
// intermediates (qkv, a1, a3) are stored bf16 in HBM (halves dominant traffic
// on a 23.3 TB/s part); bf16 operands are staged to LDS with
// global_load_async_to_lds_b128 (ASYNCcnt), fp32 operands with b128 loads +
// in-register bf16 conversion.
// ---------------------------------------------------------------------------

typedef __attribute__((ext_vector_type(16))) __bf16       v16bf;
typedef __attribute__((ext_vector_type(8)))  float        v8f;
typedef __attribute__((ext_vector_type(8)))  unsigned int v8u;

// Problem constants (reference setup: N=10000 -> H=W=100, add=0)
constexpr int D     = 768;
constexpr int QKVD  = 2304;
constexpr int NTOK  = 10001;   // cls + 10000
constexpr int PADR  = 367;     // (384 - 10001 % 384) % 384
constexpr int NP    = 10368;   // NTOK + PADR
constexpr int HEADS = 8;
constexpr int DH    = 96;
constexpr int MM    = 384;     // landmarks
constexpr int LSEG  = 27;      // NP / MM
constexpr int GW    = 100;     // grid width/height for PPEG
constexpr int NPOS  = 10000;
constexpr float QSCALE = 0.10206207261596577f;  // 96^-0.5

#define BM 64
#define BN 128
#define BK 32
#define LDSK 40   // LDS row stride in shorts: 80 B -> 16 B aligned rows (async b128)

// ---------------------------------------------------------------------------
__device__ __forceinline__ unsigned short f2bf(float f) {
  unsigned u = __float_as_uint(f);
  u += 0x7FFFu + ((u >> 16) & 1u);   // round-to-nearest-even
  return (unsigned short)(u >> 16);
}
__device__ __forceinline__ unsigned pack2bf(float lo, float hi) {
  return (unsigned)f2bf(lo) | ((unsigned)f2bf(hi) << 16);
}
__device__ __forceinline__ float bf2f(unsigned short u) {
  return __uint_as_float((unsigned)u << 16);
}

// CDNA5 async global->LDS copy (16 B per active lane), tracked by ASYNCcnt.
__device__ __forceinline__ void async_load_b128(unsigned lds_off, const void* gptr) {
  asm volatile("global_load_async_to_lds_b128 %0, %1, off"
               :: "v"(lds_off), "v"((unsigned long long)gptr)
               : "memory");
}
__device__ __forceinline__ void wait_async() {
  asm volatile("s_wait_asynccnt 0x0" ::: "memory");
}
__device__ __forceinline__ unsigned lds_addr32(const void* p) {
  return (unsigned)(unsigned long long)p;   // low 32 bits = LDS offset
}

__device__ __forceinline__ float block_reduce_sum(float v, float* sh) {
  int tid = threadIdx.x;
  sh[tid] = v; __syncthreads();
  for (int s = blockDim.x >> 1; s > 0; s >>= 1) {
    if (tid < s) sh[tid] += sh[tid + s];
    __syncthreads();
  }
  float r = sh[0]; __syncthreads();
  return r;
}

__device__ __forceinline__ float block_reduce_max(float v, float* sh) {
  int tid = threadIdx.x;
  sh[tid] = v; __syncthreads();
  for (int s = blockDim.x >> 1; s > 0; s >>= 1) {
    if (tid < s) sh[tid] = fmaxf(sh[tid], sh[tid + s]);
    __syncthreads();
  }
  float r = sh[0]; __syncthreads();
  return r;
}

// ---------------------------------------------------------------------------
// Generic batched GEMM:  C = alpha * (A x B) + diag*I (+ bias[n])
//   A: M x K row-major (lda); fp32 (ABF=0) or bf16 (ABF=1)
//   B: b_kmajor==0 -> N x K row-major ("NT"); ==1 -> K x N row-major ("NN")
//      fp32 (BBF=0) or bf16 (BBF=1)
//   C: M x N row-major (ldc); fp32 (CBF=0) or bf16 (CBF=1)
// Strides/ld in elements. Per-batch (grid.z) strides sA/sB/sC.
// Tile 64x128x32, 8 waves; each wave: one A fragment x 4 WMMA per k-step.
// bf16 operands are staged with global_load_async_to_lds_b128 (row layout in
// HBM == fragment pair layout in LDS); fp32 operands convert during staging.
// ---------------------------------------------------------------------------
template <int ABF, int BBF, int CBF>
__global__ __launch_bounds__(256) void gemm_wmma_t(
    const void* __restrict__ Av, const void* __restrict__ Bv,
    void* __restrict__ Cv, const float* __restrict__ bias,
    int Mdim, int Ndim, int Kdim, int lda, int ldb, int ldc,
    long long sA, long long sB, long long sC,
    float alpha, float diag, int b_kmajor)
{
  __shared__ __align__(16) unsigned short shA[BM][LDSK];
  __shared__ __align__(16) unsigned short shB[BN][LDSK];

  const int bz = blockIdx.z;
  const float*          Af = (const float*)Av          + (long long)bz * sA;
  const unsigned short* Ab = (const unsigned short*)Av + (long long)bz * sA;
  const float*          Bf = (const float*)Bv          + (long long)bz * sB;
  const unsigned short* Bb = (const unsigned short*)Bv + (long long)bz * sB;
  float*          Cf = (float*)Cv          + (long long)bz * sC;
  unsigned short* Cb = (unsigned short*)Cv + (long long)bz * sC;

  const int m0 = blockIdx.y * BM;
  const int n0 = blockIdx.x * BN;
  const int tid  = threadIdx.x;
  const int lane = tid & 31;
  const int wave = tid >> 5;
  const int half = lane >> 4;
  const int l16  = lane & 15;
  const int wm = (wave & 3) * 16;    // wave row offset within block tile
  const int wn = (wave >> 2) * 64;   // wave col offset within block tile

  v8f acc[4] = {{}, {}, {}, {}};

  for (int k0 = 0; k0 < Kdim; k0 += BK) {
    __syncthreads();

    // ---------------- stage A tile (64 x 32) ----------------
    if (ABF) {
      const int ar = tid >> 2;            // 0..63
      const int aseg = (tid & 3) * 8;     // 8 bf16 = 16 B
      const int gm = m0 + ar;
      if (gm < Mdim) {
        async_load_b128(lds_addr32(&shA[ar][aseg]),
                        Ab + (long long)gm * lda + (k0 + aseg));
      } else {
        unsigned* dst = (unsigned*)&shA[ar][aseg];
        dst[0] = 0u; dst[1] = 0u; dst[2] = 0u; dst[3] = 0u;
      }
    } else {
      const int sr = tid >> 2;
      const int scol = (tid & 3) * 8;
      unsigned* dst = (unsigned*)&shA[sr][scol];
      const int gm = m0 + sr;
      if (gm < Mdim) {
        const float* ap = Af + (long long)gm * lda + (k0 + scol);
        float4 v0 = *(const float4*)ap;
        float4 v1 = *(const float4*)(ap + 4);
        dst[0] = pack2bf(v0.x, v0.y);
        dst[1] = pack2bf(v0.z, v0.w);
        dst[2] = pack2bf(v1.x, v1.y);
        dst[3] = pack2bf(v1.z, v1.w);
        __builtin_prefetch(ap + BK, 0, 3);
      } else {
        dst[0] = 0u; dst[1] = 0u; dst[2] = 0u; dst[3] = 0u;
      }
    }

    // ---------------- stage B tile (128 x 32 as [n][k]) ----------------
    if (b_kmajor) {
      // B is K x N row-major: load coalesced along n, transpose into LDS.
      const int kp = tid >> 4;            // k-pair 0..15
      const int bn = (tid & 15) * 8;      // n-group of 8
      const int gn = n0 + bn;             // Ndim is a multiple of 8
      if (BBF) {
        if (gn < Ndim) {
          const unsigned short* r0 = Bb + (long long)(k0 + 2 * kp) * ldb + gn;
          const unsigned short* r1 = r0 + ldb;
          uint4 a = *(const uint4*)r0;    // 8 bf16, row k   (pairs along n)
          uint4 b = *(const uint4*)r1;    // 8 bf16, row k+1
          unsigned la[4] = {a.x, a.y, a.z, a.w};
          unsigned lb[4] = {b.x, b.y, b.z, b.w};
#pragma unroll
          for (int i = 0; i < 4; ++i) {
            *(unsigned*)&shB[bn + 2 * i][2 * kp]     = (la[i] & 0xFFFFu) | (lb[i] << 16);
            *(unsigned*)&shB[bn + 2 * i + 1][2 * kp] = (la[i] >> 16) | (lb[i] & 0xFFFF0000u);
          }
        } else {
#pragma unroll
          for (int i = 0; i < 8; ++i) *(unsigned*)&shB[bn + i][2 * kp] = 0u;
        }
      } else {
        if (gn < Ndim) {
          const float* r0 = Bf + (long long)(k0 + 2 * kp) * ldb + gn;
          const float* r1 = r0 + ldb;
          float4 l0 = *(const float4*)r0;
          float4 l1 = *(const float4*)(r0 + 4);
          float4 h0 = *(const float4*)r1;
          float4 h1 = *(const float4*)(r1 + 4);
          float lo[8] = {l0.x, l0.y, l0.z, l0.w, l1.x, l1.y, l1.z, l1.w};
          float hi[8] = {h0.x, h0.y, h0.z, h0.w, h1.x, h1.y, h1.z, h1.w};
#pragma unroll
          for (int i = 0; i < 8; ++i)
            *(unsigned*)&shB[bn + i][2 * kp] = pack2bf(lo[i], hi[i]);
          __builtin_prefetch(r0 + (long long)BK * ldb, 0, 3);
        } else {
#pragma unroll
          for (int i = 0; i < 8; ++i) *(unsigned*)&shB[bn + i][2 * kp] = 0u;
        }
      }
    } else if (BBF) {
      // B is N x K row-major bf16: async copy, two rows per thread.
      const int br = tid >> 2;            // 0..63 -> rows br and br+64
      const int bseg = (tid & 3) * 8;
#pragma unroll
      for (int rr = 0; rr < 2; ++rr) {
        const int row = br + rr * 64;
        const int gn = n0 + row;
        if (gn < Ndim) {
          async_load_b128(lds_addr32(&shB[row][bseg]),
                          Bb + (long long)gn * ldb + (k0 + bseg));
        } else {
          unsigned* dst = (unsigned*)&shB[row][bseg];
          dst[0] = 0u; dst[1] = 0u; dst[2] = 0u; dst[3] = 0u;
        }
      }
    } else {
      // B is N x K row-major fp32: b128 loads + convert.
      const int sr = tid >> 1;            // 0..127
      const int kc = (tid & 1) * 16;
      unsigned* dst = (unsigned*)&shB[sr][kc];
      const int gn = n0 + sr;
      if (gn < Ndim) {
        const float* bp = Bf + (long long)gn * ldb + (k0 + kc);
        float4 v0 = *(const float4*)bp;
        float4 v1 = *(const float4*)(bp + 4);
        float4 v2 = *(const float4*)(bp + 8);
        float4 v3 = *(const float4*)(bp + 12);
        dst[0] = pack2bf(v0.x, v0.y);
        dst[1] = pack2bf(v0.z, v0.w);
        dst[2] = pack2bf(v1.x, v1.y);
        dst[3] = pack2bf(v1.z, v1.w);
        dst[4] = pack2bf(v2.x, v2.y);
        dst[5] = pack2bf(v2.z, v2.w);
        dst[6] = pack2bf(v3.x, v3.y);
        dst[7] = pack2bf(v3.z, v3.w);
        __builtin_prefetch(bp + BK, 0, 3);
      } else {
#pragma unroll
        for (int i = 0; i < 8; ++i) dst[i] = 0u;
      }
    }

    if (ABF || BBF) wait_async();   // this wave's async copies landed
    __syncthreads();                // all waves' staging visible

    // Fragments per the documented 16-bit 16x32 layout:
    // lane half selects K-group {0..7,16..23} vs {8..15,24..31}, pairs packed.
    v8u au;
    const int am = wm + l16;
    const int kb = half * 8;
#pragma unroll
    for (int j = 0; j < 8; ++j) {
      const int kk = ((j & 4) << 2) + kb + 2 * (j & 3);
      au[j] = *(const unsigned*)&shA[am][kk];
    }
    v16bf af = __builtin_bit_cast(v16bf, au);
#pragma unroll
    for (int t = 0; t < 4; ++t) {
      v8u bu;
      const int bnr = wn + t * 16 + l16;
#pragma unroll
      for (int j = 0; j < 8; ++j) {
        const int kk = ((j & 4) << 2) + kb + 2 * (j & 3);
        bu[j] = *(const unsigned*)&shB[bnr][kk];
      }
      v16bf bf = __builtin_bit_cast(v16bf, bu);
      acc[t] = __builtin_amdgcn_wmma_f32_16x16x32_bf16(false, af, false, bf,
                                                       (short)0, acc[t], false, false);
    }
  }

  // Epilogue per 32-bit C/D layout: VGPR r -> (M = r + half*8, N = lane%16)
#pragma unroll
  for (int t = 0; t < 4; ++t) {
#pragma unroll
    for (int r = 0; r < 8; ++r) {
      const int gm = m0 + wm + r + half * 8;
      const int gn = n0 + wn + t * 16 + l16;
      if (gm < Mdim && gn < Ndim) {
        float v = alpha * acc[t][r];
        if (gm == gn) v += diag;
        if (bias) v += bias[gn];
        if (CBF) Cb[(long long)gm * ldc + gn] = f2bf(v);
        else     Cf[(long long)gm * ldc + gn] = v;
      }
    }
  }
}

// ---------------------------------------------------------------------------
// Elementwise / reduction kernels
// ---------------------------------------------------------------------------
__global__ void build_h_kernel(const float* __restrict__ hin,
                               const float* __restrict__ cls,
                               float* __restrict__ h, long long n) {
  long long i = (long long)blockIdx.x * blockDim.x + threadIdx.x;
  if (i >= n) return;
  long long row = i / D, c = i % D;
  h[i] = (row == 0) ? cls[c] : hin[(row - 1) * D + c];
}

// out[row] = 0 for row < pad, else LayerNorm(x[row - pad])
__global__ __launch_bounds__(256) void ln_pad_kernel(
    const float* __restrict__ x, float* __restrict__ out,
    const float* __restrict__ g, const float* __restrict__ b, int pad) {
  __shared__ float sh[256];
  int row = blockIdx.x;
  float* o = out + (long long)row * D;
  if (row < pad) {
    for (int c = threadIdx.x; c < D; c += 256) o[c] = 0.f;
    return;
  }
  const float* xr = x + (long long)(row - pad) * D;
  float s = 0.f, s2 = 0.f;
  for (int c = threadIdx.x; c < D; c += 256) { float v = xr[c]; s += v; s2 += v * v; }
  float ts  = block_reduce_sum(s, sh);
  float ts2 = block_reduce_sum(s2, sh);
  float mu  = ts / (float)D;
  float var = ts2 / (float)D - mu * mu;
  float inv = rsqrtf(var + 1e-5f);
  for (int c = threadIdx.x; c < D; c += 256)
    o[c] = (xr[c] - mu) * inv * g[c] + b[c];
}

__global__ __launch_bounds__(256) void softmax_rows_kernel(
    float* __restrict__ mat, int cols, long long ld) {
  __shared__ float sh[256];
  float* p = mat + (long long)blockIdx.x * ld;
  float m = -3.402823466e38f;
  for (int c = threadIdx.x; c < cols; c += 256) m = fmaxf(m, p[c]);
  m = block_reduce_max(m, sh);
  float s = 0.f;
  for (int c = threadIdx.x; c < cols; c += 256) {
    float e = __expf(p[c] - m);
    p[c] = e; s += e;
  }
  s = block_reduce_sum(s, sh);
  float inv = 1.f / s;
  for (int c = threadIdx.x; c < cols; c += 256) p[c] *= inv;
}

// in-place row softmax on a bf16 matrix (math in f32)
__global__ __launch_bounds__(256) void softmax_rows_bf16_kernel(
    unsigned short* __restrict__ mat, int cols, long long ld) {
  __shared__ float sh[256];
  unsigned short* p = mat + (long long)blockIdx.x * ld;
  float m = -3.402823466e38f;
  for (int c = threadIdx.x; c < cols; c += 256) m = fmaxf(m, bf2f(p[c]));
  m = block_reduce_max(m, sh);
  float s = 0.f;
  for (int c = threadIdx.x; c < cols; c += 256) {
    float e = __expf(bf2f(p[c]) - m);
    p[c] = f2bf(e); s += e;
  }
  s = block_reduce_sum(s, sh);
  float inv = 1.f / s;
  for (int c = threadIdx.x; c < cols; c += 256)
    p[c] = f2bf(bf2f(p[c]) * inv);
}

// Landmarks: ql (q-scaled), kl = mean over 27-token segments of q/k (bf16 qkv)
__global__ void meanpool_kernel(const unsigned short* __restrict__ qkv,
                                float* __restrict__ ql, float* __restrict__ kl) {
  int m = blockIdx.x, h = blockIdx.y, d = threadIdx.x;
  float sq = 0.f, sk = 0.f;
  for (int l = 0; l < LSEG; ++l) {
    long long row = (long long)(m * LSEG + l) * QKVD;
    sq += bf2f(qkv[row + h * DH + d]);
    sk += bf2f(qkv[row + D + h * DH + d]);
  }
  float inv = 1.f / (float)LSEG;
  long long o = ((long long)h * MM + m) * DH + d;
  ql[o] = sq * inv * QSCALE;
  kl[o] = sk * inv;
}

__global__ void zero_kernel(float* p, int n) {
  int i = blockIdx.x * blockDim.x + threadIdx.x;
  if (i < n) p[i] = 0.f;
}

// atomicMax over non-negative floats via int-bit compare (softmax sums >= 0)
__global__ __launch_bounds__(128) void rowsum_max_kernel(
    const float* __restrict__ x, float* out) {
  __shared__ float sh[128];
  const float* p = x + (long long)blockIdx.x * MM;
  float s = 0.f;
  for (int c = threadIdx.x; c < MM; c += 128) s += fabsf(p[c]);
  s = block_reduce_sum(s, sh);
  if (threadIdx.x == 0) atomicMax((int*)out, __float_as_int(s));
}

__global__ __launch_bounds__(128) void colsum_max_kernel(
    const float* __restrict__ x, float* out) {
  __shared__ float sh[128];
  int h = blockIdx.x / MM, j = blockIdx.x % MM;
  const float* p = x + (long long)h * MM * MM + j;
  float s = 0.f;
  for (int i = threadIdx.x; i < MM; i += 128) s += fabsf(p[(long long)i * MM]);
  s = block_reduce_sum(s, sh);
  if (threadIdx.x == 0) atomicMax((int*)out, __float_as_int(s));
}

// z0 = x^T / (rowmax * colmax)
__global__ void transpose_scale_kernel(const float* __restrict__ x,
                                       float* __restrict__ z,
                                       const float* __restrict__ sc) {
  int h = blockIdx.z;
  int i = blockIdx.y * 16 + threadIdx.y;
  int j = blockIdx.x * 16 + threadIdx.x;
  float inv = 1.f / (sc[0] * sc[1]);
  z[((long long)h * MM + i) * MM + j] = x[((long long)h * MM + j) * MM + i] * inv;
}

// merged[i][h*96+d] += depthwise conv33 over tokens of v (bf16, zero padded 16)
__global__ __launch_bounds__(96) void resconv_add_kernel(
    const unsigned short* __restrict__ qkv, const float* __restrict__ rw,
    float* __restrict__ merged) {
  int i = blockIdx.x, h = blockIdx.y, d = threadIdx.x;
  float s = 0.f;
#pragma unroll
  for (int k = 0; k < 33; ++k) {
    int j = i + k - 16;
    if (j >= 0 && j < NP)
      s += bf2f(qkv[(long long)j * QKVD + 2 * D + h * DH + d]) * rw[h * 33 + k];
  }
  merged[(long long)i * D + h * DH + d] += s;
}

// h[t] += y[t + PADR]  (keep last NTOK rows of the padded projection)
__global__ void residual_add_kernel(float* __restrict__ h,
                                    const float* __restrict__ y, long long n) {
  long long i = (long long)blockIdx.x * blockDim.x + threadIdx.x;
  if (i >= n) return;
  long long t = i / D, c = i % D;
  h[i] += y[(t + PADR) * D + c];
}

// f[c*NPOS + pos] = h[(1+pos)*D + c]  (channel-major feature map)
__global__ void ppeg_copy_kernel(const float* __restrict__ h,
                                 float* __restrict__ f, long long n) {
  long long i = (long long)blockIdx.x * blockDim.x + threadIdx.x;
  if (i >= n) return;
  long long c = i / NPOS, pos = i % NPOS;
  f[i] = h[(1 + pos) * D + c];
}

// h[1+pos] = f + dw7(f) + dw5(f) + dw3(f) + biases   (100x100 grid, zero pad)
__global__ __launch_bounds__(256) void ppeg_conv_kernel(
    const float* __restrict__ f, float* __restrict__ h,
    const float* __restrict__ w7, const float* __restrict__ b7,
    const float* __restrict__ w5, const float* __restrict__ b5,
    const float* __restrict__ w3, const float* __restrict__ b3) {
  int pos = blockIdx.x;
  int row = pos / GW, col = pos % GW;
  for (int c = threadIdx.x; c < D; c += 256) {
    const float* fc = f + (long long)c * NPOS;
    float s = fc[pos];
    const float* W7 = w7 + c * 49;
    for (int ky = 0; ky < 7; ++ky) {
      int yy = row + ky - 3;
      if (yy < 0 || yy >= GW) continue;
      for (int kx = 0; kx < 7; ++kx) {
        int xx = col + kx - 3;
        if (xx < 0 || xx >= GW) continue;
        s += fc[yy * GW + xx] * W7[ky * 7 + kx];
      }
    }
    const float* W5 = w5 + c * 25;
    for (int ky = 0; ky < 5; ++ky) {
      int yy = row + ky - 2;
      if (yy < 0 || yy >= GW) continue;
      for (int kx = 0; kx < 5; ++kx) {
        int xx = col + kx - 2;
        if (xx < 0 || xx >= GW) continue;
        s += fc[yy * GW + xx] * W5[ky * 5 + kx];
      }
    }
    const float* W3 = w3 + c * 9;
    for (int ky = 0; ky < 3; ++ky) {
      int yy = row + ky - 1;
      if (yy < 0 || yy >= GW) continue;
      for (int kx = 0; kx < 3; ++kx) {
        int xx = col + kx - 1;
        if (xx < 0 || xx >= GW) continue;
        s += fc[yy * GW + xx] * W3[ky * 3 + kx];
      }
    }
    s += b7[c] + b5[c] + b3[c];
    h[(long long)(1 + pos) * D + c] = s;
  }
}

// logits / softmax / argmax for 16 heads x 2 classes; hc at out+80
__global__ __launch_bounds__(256) void classifier_kernel(
    const float* __restrict__ hc, const float* __restrict__ cw,
    const float* __restrict__ cb, float* __restrict__ out) {
  __shared__ float sh[256];
  int k = blockIdx.x;
  float s0 = 0.f, s1 = 0.f;
  for (int d = threadIdx.x; d < D; d += 256) {
    float v = hc[d];
    s0 += v * cw[((long long)k * 2 + 0) * D + d];
    s1 += v * cw[((long long)k * 2 + 1) * D + d];
  }
  float t0 = block_reduce_sum(s0, sh);
  float t1 = block_reduce_sum(s1, sh);
  if (threadIdx.x == 0) {
    float l0 = t0 + cb[k * 2 + 0];
    float l1 = t1 + cb[k * 2 + 1];
    out[k * 2 + 0] = l0;
    out[k * 2 + 1] = l1;
    float mx = fmaxf(l0, l1);
    float e0 = __expf(l0 - mx), e1 = __expf(l1 - mx);
    float inv = 1.f / (e0 + e1);
    out[32 + k * 2 + 0] = e0 * inv;
    out[32 + k * 2 + 1] = e1 * inv;
    out[64 + k] = (l1 > l0) ? 1.f : 0.f;  // argmax, first-index on tie
  }
}

// ---------------------------------------------------------------------------
// Host-side orchestration
// ---------------------------------------------------------------------------
template <int ABF, int BBF, int CBF>
static void launch_gemm(hipStream_t st, const void* A, const void* B, void* C,
                        const float* bias, int M, int N, int K,
                        int lda, int ldb, int ldc,
                        long long sA, long long sB, long long sC, int batch,
                        float alpha, float diag, int bkmaj) {
  dim3 g((N + BN - 1) / BN, (M + BM - 1) / BM, batch);
  gemm_wmma_t<ABF, BBF, CBF><<<g, 256, 0, st>>>(A, B, C, bias, M, N, K,
                                                lda, ldb, ldc, sA, sB, sC,
                                                alpha, diag, bkmaj);
}

static void run_layer(hipStream_t st, float* wsH, float* wsXP, float* wsQKV,
                      float* wsQL, float* wsKL, float* wsA1, float* wsA3,
                      float* wsPX, float* wsZA, float* wsZB, float* wsXZ,
                      float* wsT2, float* wsT4, float* wsA3V, float* wsSC,
                      const float* lng, const float* lnb, const float* qkvw,
                      const float* outw, const float* outb, const float* resw) {
  const long long sMM  = (long long)MM * MM;
  const long long sMD  = (long long)MM * DH;
  const long long sNPM = (long long)NP * MM;

  unsigned short* qkb = (unsigned short*)wsQKV;   // qkv lives as bf16
  unsigned short* a1b = (unsigned short*)wsA1;    // a1 as bf16
  unsigned short* a3b = (unsigned short*)wsA3;    // a3 / t as bf16

  // LayerNorm + front pad -> xp (NP x D, f32)
  ln_pad_kernel<<<NP, 256, 0, st>>>(wsH, wsXP, lng, lnb, PADR);
  // qkv = xp @ qkv_w^T  -> bf16 (NP x 2304)
  launch_gemm<0, 0, 1>(st, wsXP, qkvw, qkb, nullptr, NP, QKVD, D, D, D, QKVD,
                       0, 0, 0, 1, 1.f, 0.f, 0);
  // landmarks ql (scaled), kl (f32)
  meanpool_kernel<<<dim3(MM, HEADS), DH, 0, st>>>(qkb, wsQL, wsKL);
  // a1 = softmax(scale * q @ kl^T) -> bf16   (per head: NP x 384)
  launch_gemm<1, 0, 1>(st, qkb, wsKL, a1b, nullptr, NP, MM, DH, QKVD, DH, MM,
                       DH, sMD, sNPM, HEADS, QSCALE, 0.f, 0);
  softmax_rows_bf16_kernel<<<HEADS * NP, 256, 0, st>>>(a1b, MM, MM);
  // a2 = softmax(ql @ kl^T) -> px (f32, pinv wants precision)
  launch_gemm<0, 0, 0>(st, wsQL, wsKL, wsPX, nullptr, MM, MM, DH, DH, DH, MM,
                       sMD, sMD, sMM, HEADS, 1.f, 0.f, 0);
  softmax_rows_kernel<<<HEADS * MM, 256, 0, st>>>(wsPX, MM, MM);
  // a3 = softmax(ql @ k^T) -> bf16  (per head: 384 x NP; B async-staged)
  launch_gemm<0, 1, 1>(st, wsQL, qkb + D, a3b, nullptr, MM, NP, DH, DH, QKVD, NP,
                       sMD, DH, (long long)MM * NP, HEADS, 1.f, 0.f, 0);
  softmax_rows_bf16_kernel<<<HEADS * MM, 256, 0, st>>>(a3b, NP, NP);
  // pinv init: z = x^T / (max rowsum * max colsum)
  zero_kernel<<<1, 32, 0, st>>>(wsSC, 2);
  rowsum_max_kernel<<<HEADS * MM, 128, 0, st>>>(wsPX, wsSC);
  colsum_max_kernel<<<HEADS * MM, 128, 0, st>>>(wsPX, wsSC + 1);
  transpose_scale_kernel<<<dim3(MM / 16, MM / 16, HEADS), dim3(16, 16), 0, st>>>(
      wsPX, wsZA, wsSC);
  // 6 Newton-Schulz iterations (all f32): z = 0.25*z@(13I - xz@(7I - xz@xz))
  float* za = wsZA; float* zb = wsZB;
  for (int it = 0; it < 6; ++it) {
    launch_gemm<0, 0, 0>(st, wsPX, za,   wsXZ, nullptr, MM, MM, MM, MM, MM, MM,
                         sMM, sMM, sMM, HEADS, 1.f, 0.f, 1);
    launch_gemm<0, 0, 0>(st, wsXZ, wsXZ, wsT2, nullptr, MM, MM, MM, MM, MM, MM,
                         sMM, sMM, sMM, HEADS, -1.f, 7.f, 1);
    launch_gemm<0, 0, 0>(st, wsXZ, wsT2, wsT4, nullptr, MM, MM, MM, MM, MM, MM,
                         sMM, sMM, sMM, HEADS, -1.f, 13.f, 1);
    launch_gemm<0, 0, 0>(st, za,   wsT4, zb,   nullptr, MM, MM, MM, MM, MM, MM,
                         sMM, sMM, sMM, HEADS, 0.25f, 0.f, 1);
    float* t = za; za = zb; zb = t;
  }
  // a3v = a3 @ v   (A async bf16, B bf16 k-major; per head: 384 x 96, f32 out)
  launch_gemm<1, 1, 0>(st, a3b, qkb + 2 * D, wsA3V, nullptr, MM, DH, NP,
                       NP, QKVD, DH, (long long)MM * NP, DH, sMD, HEADS,
                       1.f, 0.f, 1);
  // t = a1 @ pinv(a2) -> bf16 (reuse a3 buffer; per head: NP x 384)
  launch_gemm<1, 0, 1>(st, a1b, za, a3b, nullptr, NP, MM, MM, MM, MM, MM,
                       sNPM, sMM, sNPM, HEADS, 1.f, 0.f, 1);
  // merged[i][h*96+d] = t @ a3v  (f32 out into xp buffer, head-interleaved)
  launch_gemm<1, 0, 0>(st, a3b, wsA3V, wsXP, nullptr, NP, DH, MM, MM, DH, D,
                       sNPM, sMD, DH, HEADS, 1.f, 0.f, 1);
  // + depthwise residual conv of v (bf16)
  resconv_add_kernel<<<dim3(NP, HEADS), DH, 0, st>>>(qkb, resw, wsXP);
  // projection: y = merged @ out_w^T + out_b  (f32, reuses qkv region)
  launch_gemm<0, 0, 0>(st, wsXP, outw, wsQKV, outb, NP, D, D, D, D, D,
                       0, 0, 0, 1, 1.f, 0.f, 0);
  // h += y[-NTOK:]
  long long n = (long long)NTOK * D;
  residual_add_kernel<<<(unsigned)((n + 255) / 256), 256, 0, st>>>(wsH, wsQKV, n);
}

extern "C" void kernel_launch(void* const* d_in, const int* in_sizes, int n_in,
                              void* d_out, int out_size, void* d_ws, size_t ws_size,
                              hipStream_t stream) {
  (void)in_sizes; (void)n_in; (void)out_size; (void)ws_size;
  const float* h_in  = (const float*)d_in[0];
  const float* clsT  = (const float*)d_in[1];
  const float* ln1g  = (const float*)d_in[2];
  const float* ln1b  = (const float*)d_in[3];
  const float* qkv1w = (const float*)d_in[4];
  const float* out1w = (const float*)d_in[5];
  const float* out1b = (const float*)d_in[6];
  const float* res1w = (const float*)d_in[7];
  const float* ln2g  = (const float*)d_in[8];
  const float* ln2b  = (const float*)d_in[9];
  const float* qkv2w = (const float*)d_in[10];
  const float* out2w = (const float*)d_in[11];
  const float* out2b = (const float*)d_in[12];
  const float* res2w = (const float*)d_in[13];
  const float* c7w   = (const float*)d_in[14];
  const float* c7b   = (const float*)d_in[15];
  const float* c5w   = (const float*)d_in[16];
  const float* c5b   = (const float*)d_in[17];
  const float* c3w   = (const float*)d_in[18];
  const float* c3b   = (const float*)d_in[19];
  const float* lnfg  = (const float*)d_in[20];
  const float* lnfb  = (const float*)d_in[21];
  const float* clsw  = (const float*)d_in[22];
  const float* clsb  = (const float*)d_in[23];

  // Workspace layout (regions sized in f32 elements; bf16 users take half):
  float* ws    = (float*)d_ws;
  float* wsH   = ws;                                   // NTOK*D  (state, f32)
  float* wsXP  = wsH   + (long long)NTOK * D;          // NP*D    (xp / merged, f32)
  float* wsQKV = wsXP  + (long long)NP * D;            // NP*QKVD (qkv bf16 / y f32 / ppeg feat f32)
  float* wsQL  = wsQKV + (long long)NP * QKVD;         // H*MM*DH f32
  float* wsKL  = wsQL  + (long long)HEADS * MM * DH;   // H*MM*DH f32
  float* wsA1  = wsKL  + (long long)HEADS * MM * DH;   // H*NP*MM (bf16)
  float* wsA3  = wsA1  + (long long)HEADS * NP * MM;   // H*NP*MM (a3 / t, bf16)
  float* wsPX  = wsA3  + (long long)HEADS * NP * MM;   // H*MM*MM f32 (a2)
  float* wsZA  = wsPX  + (long long)HEADS * MM * MM;
  float* wsZB  = wsZA  + (long long)HEADS * MM * MM;
  float* wsXZ  = wsZB  + (long long)HEADS * MM * MM;
  float* wsT2  = wsXZ  + (long long)HEADS * MM * MM;
  float* wsT4  = wsT2  + (long long)HEADS * MM * MM;
  float* wsA3V = wsT4  + (long long)HEADS * MM * MM;   // H*MM*DH f32
  float* wsSC  = wsA3V + (long long)HEADS * MM * DH;   // 2 scalars

  // h = [cls ; h_in]  (add = H*H - N = 0 for N=10000)
  long long nH = (long long)NTOK * D;
  build_h_kernel<<<(unsigned)((nH + 255) / 256), 256, 0, stream>>>(h_in, clsT, wsH, nH);

  // Layer 1
  run_layer(stream, wsH, wsXP, wsQKV, wsQL, wsKL, wsA1, wsA3, wsPX, wsZA, wsZB,
            wsXZ, wsT2, wsT4, wsA3V, wsSC, ln1g, ln1b, qkv1w, out1w, out1b, res1w);

  // PPEG (cls token untouched); feature buffer reuses qkv region (f32)
  long long nF = (long long)D * NPOS;
  ppeg_copy_kernel<<<(unsigned)((nF + 255) / 256), 256, 0, stream>>>(wsH, wsQKV, nF);
  ppeg_conv_kernel<<<NPOS, 256, 0, stream>>>(wsQKV, wsH, c7w, c7b, c5w, c5b, c3w, c3b);

  // Layer 2
  run_layer(stream, wsH, wsXP, wsQKV, wsQL, wsKL, wsA1, wsA3, wsPX, wsZA, wsZB,
            wsXZ, wsT2, wsT4, wsA3V, wsSC, ln2g, ln2b, qkv2w, out2w, out2b, res2w);

  // Final: hc = LN(h)[0] -> d_out[80..848); then logits/probs/preds
  float* outF = (float*)d_out;
  ln_pad_kernel<<<1, 256, 0, stream>>>(wsH, outF + 80, lnfg, lnfb, 0);
  classifier_kernel<<<16, 256, 0, stream>>>(outF + 80, clsw, clsb, outF);
}
